// WKA_29051158790344
// MI455X (gfx1250) — matrix operand order repaired and checked by reading, not compiled, not run on hardware
//
#include <hip/hip_runtime.h>
#include <hip/hip_bf16.h>
#include <math.h>

// ---------------------------------------------------------------------------
// SAR omega-K (WKA) pipeline for MI455X (gfx1250, wave32, WMMA + TDM).
// f32 complex data path, f64 phase generation (mod-2pi reduced), radix-16
// FFT stage on V_WMMA_F32_16X16X32_F16, row fills via TENSOR_LOAD_TO_LDS.
// ---------------------------------------------------------------------------

typedef __attribute__((ext_vector_type(16))) _Float16 v16h;
typedef __attribute__((ext_vector_type(8)))  float    v8f;
typedef __attribute__((ext_vector_type(4)))  unsigned u32x4;
typedef __attribute__((ext_vector_type(8)))  int      i32x8;
typedef __attribute__((ext_vector_type(4)))  int      i32x4;

#define N2048 2048
#define NNTOT (2048ull * 2048ull)

// TDM availability; this toolchain exposes the 6-arg (clang-23 / therock)
// form: (u32x4 g0, i32x8 g1, i32x4, i32x4, i32x8, i32 cpol). Guarded so the
// file still compiles on the host pass / toolchains without the builtin.
#if defined(__HIP_DEVICE_COMPILE__) &&                                   \
    __has_builtin(__builtin_amdgcn_tensor_load_to_lds) &&                \
    __has_builtin(__builtin_amdgcn_s_wait_tensorcnt)
#define WKA_HAVE_TDM 1
#else
#define WKA_HAVE_TDM 0
#endif

// --- SAR constants (double precision) ---
constexpr double D_PI      = 3.14159265358979323846;
constexpr double D_TWOPI   = 6.28318530717958647693;
constexpr double D_VR      = 7100.0;
constexpr double D_THETAC  = 0.06;
constexpr double D_C       = 3.0e8;
constexpr double D_F0      = 5.3e9;
constexpr double D_LAMDA   = D_C / D_F0;
constexpr double D_FR      = 60.0e6;
constexpr double D_FA      = 1700.0;
constexpr double D_RREF    = 850.0e3;
constexpr double D_KR      = 0.72e12;

__device__ __forceinline__ float reduce_2pi(double ph) {
  // large-magnitude phase -> [0, 2pi) in double, then safe for f32 sincos
  double t = ph - D_TWOPI * floor(ph * (1.0 / D_TWOPI));
  return (float)t;
}

__device__ __forceinline__ float sinc_pi(float x) {
  float ax = fabsf(x);
  if (ax < 1e-6f) return 1.0f;
  float px = (float)D_PI * x;
  return __sinf(px) / px;
}

#if WKA_HAVE_TDM
// raw 32-bit LDS byte offset of a __shared__ object (generic -> AS(3) -> int)
__device__ __forceinline__ unsigned wka_lds_offset(const void* p) {
  return (unsigned)(unsigned long long)(__attribute__((address_space(3))) const void*)p;
}
#endif

// ---------------------------------------------------------------------------
// pack: (real, imag) -> interleaved complex f32
// ---------------------------------------------------------------------------
__global__ __launch_bounds__(256) void pack_kernel(const float* __restrict__ re,
                                                   const float* __restrict__ im,
                                                   float2* __restrict__ dst) {
  size_t i = (size_t)blockIdx.x * 256 + threadIdx.x;
  dst[i] = make_float2(re[i], im[i]);
}

// ---------------------------------------------------------------------------
// 2048-pt FFT per row, one workgroup (8 wave32) per row, fully in LDS.
// Four-step:  n = n2 + 128*n1, k = k1 + 16*k2
//   fill:    TDM 1-D tile copy (global row -> LDS), TENSORcnt tracked
//   stage A: DFT-16 over n1  -> complex 16x128 matmul on WMMA (2 ops / tile)
//   twiddle: W_2048^{k1*n2}
//   stage B: DFT-128 over n2 -> 7-pass radix-2 Stockham in LDS
// SHIFT folds ifftshift (stage-A gather) / fftshift (store) = roll by 1024.
// DIR = -1 forward, +1 inverse; SCALE multiplies by 1/2048 (inverse).
// ---------------------------------------------------------------------------
template <int DIR, bool SHIFT, bool SCALE>
__global__ __launch_bounds__(256) void fft2048_rows(const float2* __restrict__ in,
                                                    float2* __restrict__ out) {
  __shared__ float2 bufA[N2048];
  __shared__ float2 bufB[N2048];

  const int row  = blockIdx.x;
  const int tid  = threadIdx.x;
  const int lane = tid & 31;
  const int wave = tid >> 5;
  const float2* src = in + (size_t)row * N2048;

#if WKA_HAVE_TDM
  // ---- row fill via Tensor Data Mover -------------------------------------
  if (wave == 0) {
    unsigned long long ga = (unsigned long long)(const void*)src;
    unsigned lds = wka_lds_offset(&bufA[0]);
    // D# group0: count=1 | lds_addr | global_addr[31:0] | ga[56:32], type=2
    u32x4 g0 = { 1u, lds, (unsigned)(ga & 0xFFFFFFFFull),
                 (unsigned)((ga >> 32) & 0x01FFFFFFull) | (2u << 30) };
    // D# group1: data_size=8B; tensor_dim0=2048; tensor_dim1=1;
    //            tile_dim0=2048; tile_dim1=1; tensor_dim0_stride=2048
    i32x8 g1 = { (int)(3u << 16),          // workgroup_mask=0, data_size=3 (8B)
                 (int)(2048u << 16),       // tensor_dim0[15:0] at bit 48
                 (int)(1u << 16),          // tensor_dim0 hi=0, tensor_dim1=1
                 (int)(2048u << 16),       // tensor_dim1 hi=0, tile_dim0=2048
                 1,                        // tile_dim1=1, tile_dim2=0
                 2048,                     // tensor_dim0_stride[31:0]
                 0, 0 };
    i32x4 gz4 = { 0, 0, 0, 0 };
    i32x8 gz8 = { 0, 0, 0, 0, 0, 0, 0, 0 };
    __builtin_amdgcn_tensor_load_to_lds(g0, g1, gz4, gz4, gz8, 0);
    __builtin_amdgcn_s_wait_tensorcnt(0);
  }
  __syncthreads();
#else
  for (int i = tid; i < N2048; i += 256) bufA[i] = src[i];
  __syncthreads();
#endif

  // ---- stage A: radix-16 via WMMA -----------------------------------------
  const int m  = lane & 15;   // output row k1 base / B column
  const int hi = lane >> 4;

  // A matrices (16x32 f16): A_re = [cos | -DIR*sin], A_im = [DIR*sin | cos]
  v16h Are, Aim;
  #pragma unroll
  for (int e = 0; e < 16; e++) {
    int K  = 16 * hi + e;
    int n1 = K & 15;
    float th = (float)(D_TWOPI / 16.0) * (float)((m * n1) & 15);
    float sv, cv;
    __sincosf(th, &sv, &cv);
    float are = (K < 16) ? cv : (-(float)DIR) * sv;
    float aim = (K < 16) ? ((float)DIR) * sv : cv;
    Are[e] = (_Float16)are;
    Aim[e] = (_Float16)aim;
  }

  {
    const int t  = wave;           // each wave owns one 16-column tile
    const int n2 = t * 16 + m;     // global n2 column

    // B (32x16 f16): rows 0..15 = Re over n1, rows 16..31 = Im over n1
    // ifftshift folded into the gather index (roll by 1024, even N).
    v16h B;
    #pragma unroll
    for (int e = 0; e < 16; e++) {
      int n  = n2 + 128 * e;
      int s  = SHIFT ? ((n + 1024) & 2047) : n;
      float2 v = bufA[s];
      B[e] = (_Float16)(hi ? v.y : v.x);
    }

    v8f c0 = {};
    v8f Dre = __builtin_amdgcn_wmma_f32_16x16x32_f16(false, Are, false, B,
                                                     (short)0, c0, false, false);
    v8f Dim = __builtin_amdgcn_wmma_f32_16x16x32_f16(false, Aim, false, B,
                                                     (short)0, c0, false, false);

    // twiddle W_2048^{k1*n2} and scatter to bufB[k1*128 + n2]
    #pragma unroll
    for (int j = 0; j < 8; j++) {
      int k1 = j + 8 * hi;
      float ph = (float)DIR * (float)(D_TWOPI / 2048.0) * (float)(k1 * n2);
      float sw, cw;
      __sincosf(ph, &sw, &cw);
      float re = Dre[j], im = Dim[j];
      bufB[k1 * 128 + n2] = make_float2(re * cw - im * sw, re * sw + im * cw);
    }
  }
  __syncthreads();

  // ---- stage B: 16 independent 128-pt Stockham radix-2 FFTs ---------------
  float2* sb = bufB;
  float2* db = bufA;
  int l = 64, mm = 1, mlog = 0;
  for (int pass = 0; pass < 7; pass++) {
    #pragma unroll
    for (int i = 0; i < 4; i++) {
      int u    = tid + 256 * i;      // 1024 butterflies / pass
      int sub  = u >> 6;             // which k1 row
      int q    = u & 63;
      int j    = q >> mlog;
      int k    = q & (mm - 1);
      int base = sub * 128;
      float2 a = sb[base + k + mm * j];
      float2 b = sb[base + k + mm * (j + l)];
      float ph = (float)DIR * (float)D_PI * (float)j / (float)l;
      float sw, cw;
      __sincosf(ph, &sw, &cw);
      float dx = a.x - b.x, dy = a.y - b.y;
      db[base + k + mm * (2 * j)]     = make_float2(a.x + b.x, a.y + b.y);
      db[base + k + mm * (2 * j + 1)] = make_float2(dx * cw - dy * sw,
                                                    dx * sw + dy * cw);
    }
    __syncthreads();
    float2* tmp = sb; sb = db; db = tmp;
    l >>= 1; mm <<= 1; mlog++;
  }

  // ---- store: k = k1 + 16*k2, fold fftshift ------------------------------
  const float scale = SCALE ? (1.0f / 2048.0f) : 1.0f;
  float2* dst = out + (size_t)row * N2048;
  for (int i = tid; i < N2048; i += 256) {
    int k1 = i >> 7, k2 = i & 127;
    int k  = k1 + (k2 << 4);
    int o  = SHIFT ? ((k + 1024) & 2047) : k;
    float2 v = sb[i];
    dst[o] = make_float2(v.x * scale, v.y * scale);
  }
}

// ---------------------------------------------------------------------------
// LDS-tiled transpose (32x32 float2 tiles)
// ---------------------------------------------------------------------------
__global__ __launch_bounds__(256) void transpose2048(const float2* __restrict__ in,
                                                     float2* __restrict__ out) {
  __shared__ float2 tile[32][33];
  int bx = blockIdx.x * 32, by = blockIdx.y * 32;
  int tx = threadIdx.x, ty = threadIdx.y;  // 32 x 8
  #pragma unroll
  for (int r = 0; r < 32; r += 8)
    tile[ty + r][tx] = in[(size_t)(by + ty + r) * N2048 + bx + tx];
  __syncthreads();
  #pragma unroll
  for (int r = 0; r < 32; r += 8)
    out[(size_t)(bx + ty + r) * N2048 + by + tx] = tile[tx][ty + r];
}

// ---------------------------------------------------------------------------
// matched filter: emit s2df, then W *= h2df, emit s2df_matched.
// Phase ~1.9e8 rad -> compute in f64, reduce mod 2pi, f32 sincos.
// ---------------------------------------------------------------------------
__global__ __launch_bounds__(256) void matched_kernel(float2* __restrict__ W,
                                                      float* __restrict__ out_s2df,
                                                      float* __restrict__ out_matched) {
  size_t i = (size_t)blockIdx.x * 256 + threadIdx.x;
  int a = (int)(i >> 11);
  int r = (int)(i & 2047);

  double f_eta_ref = 2.0 * D_VR * sin(D_THETAC) / D_LAMDA;
  double f_tau = -D_FR * 0.5 + (double)r * (D_FR / 2047.0);
  double f_eta = -D_FA * 0.5 + (double)a * (D_FA / 2047.0) + f_eta_ref;

  double t1   = D_F0 + f_tau;
  double root = sqrt(t1 * t1 - (D_C * D_C) * (f_eta * f_eta) / (4.0 * D_VR * D_VR));
  double phase = 4.0 * D_PI * D_RREF / D_C * root + D_PI * f_tau * f_tau / D_KR;

  float ph = reduce_2pi(phase);
  float sw, cw;
  __sincosf(ph, &sw, &cw);

  float2 s = W[i];
  out_s2df[2 * i]     = s.x;
  out_s2df[2 * i + 1] = s.y;

  float2 mres = make_float2(s.x * cw - s.y * sw, s.x * sw + s.y * cw);
  W[i] = mres;
  out_matched[2 * i]     = mres.x;
  out_matched[2 * i + 1] = mres.y;
}

// ---------------------------------------------------------------------------
// Stolt interpolation: 6-tap sinc gather along range + phase comp.
// f_tao_pie - f_tau is a catastrophic-cancellation term -> f64.
// ---------------------------------------------------------------------------
__global__ __launch_bounds__(256) void stolt_kernel(const float2* __restrict__ Wm,
                                                    float2* __restrict__ Wo,
                                                    float* __restrict__ out_stolt) {
  size_t i = (size_t)blockIdx.x * 256 + threadIdx.x;
  int a = (int)(i >> 11);
  int r = (int)(i & 2047);

  double f_eta_ref = 2.0 * D_VR * sin(D_THETAC) / D_LAMDA;
  double f_tau = -D_FR * 0.5 + (double)r * (D_FR / 2047.0);
  double f_eta = -D_FA * 0.5 + (double)a * (D_FA / 2047.0) + f_eta_ref;

  double t1 = D_F0 + f_tau;
  double q  = D_C * f_eta / (2.0 * D_VR);
  double f_tao_pie = sqrt(t1 * t1 - q * q) - D_F0;
  float  delta = (float)((f_tao_pie - f_tau) * (2048.0 / D_FR));

  const float2* rowp = Wm + (size_t)a * N2048;
  __builtin_prefetch(rowp + (r < 3 ? 0 : r - 3), 0, 3);   // global_prefetch_b8

  float2 acc = make_float2(0.0f, 0.0f);
  #pragma unroll
  for (int t = 0; t < 6; t++) {
    int off = t - 3;                          // offsets -3..2
    int idx = r + off;
    idx = idx < 0 ? 0 : (idx > 2047 ? 2047 : idx);
    float w  = sinc_pi(delta - (float)off);
    float2 s = rowp[idx];
    acc.x += s.x * w;
    acc.y += s.y * w;
  }

  float ph = reduce_2pi(-4.0 * D_PI * D_RREF / D_C * f_tau);
  float sw, cw;
  __sincosf(ph, &sw, &cw);
  float2 res = make_float2(acc.x * cw - acc.y * sw, acc.x * sw + acc.y * cw);

  Wo[i] = res;
  out_stolt[2 * i]     = res.x;
  out_stolt[2 * i + 1] = res.y;
}

// ---------------------------------------------------------------------------
// final image emit
// ---------------------------------------------------------------------------
__global__ __launch_bounds__(256) void emit_kernel(const float2* __restrict__ W,
                                                   float* __restrict__ out_image) {
  size_t i = (size_t)blockIdx.x * 256 + threadIdx.x;
  float2 v = W[i];
  out_image[2 * i]     = v.x;
  out_image[2 * i + 1] = v.y;
}

// ---------------------------------------------------------------------------
extern "C" void kernel_launch(void* const* d_in, const int* in_sizes, int n_in,
                              void* d_out, int out_size, void* d_ws, size_t ws_size,
                              hipStream_t stream) {
  (void)in_sizes; (void)n_in; (void)out_size; (void)ws_size;

  const float* sig_real = (const float*)d_in[0];
  const float* sig_imag = (const float*)d_in[1];

  float* out = (float*)d_out;
  float* out_s2df    = out;                   // complex interleaved
  float* out_matched = out + 2ull * NNTOT;
  float* out_stolt   = out + 4ull * NNTOT;
  float* out_image   = out + 6ull * NNTOT;

  float2* W0 = (float2*)d_ws;                 // 32 MB
  float2* W1 = W0 + NNTOT;                    // 32 MB (64 MB total, L2-resident)

  const int PW_BLOCKS = (int)(NNTOT / 256);
  dim3 tg(64, 64), tb(32, 8);

  // pack
  pack_kernel<<<PW_BLOCKS, 256, 0, stream>>>(sig_real, sig_imag, W0);

  // centered forward 2D FFT
  fft2048_rows<-1, true, false><<<N2048, 256, 0, stream>>>(W0, W0);  // range
  transpose2048<<<tg, tb, 0, stream>>>(W0, W1);
  fft2048_rows<-1, true, false><<<N2048, 256, 0, stream>>>(W1, W1);  // azimuth
  transpose2048<<<tg, tb, 0, stream>>>(W1, W0);                      // W0 = s2df

  // matched filter (emits s2df + s2df_matched); W0 <- matched
  matched_kernel<<<PW_BLOCKS, 256, 0, stream>>>(W0, out_s2df, out_matched);

  // Stolt interpolation + phase comp; W1 <- stolt (emits stolt_2df)
  stolt_kernel<<<PW_BLOCKS, 256, 0, stream>>>(W0, W1, out_stolt);

  // centered inverse 2D FFT
  fft2048_rows<1, true, true><<<N2048, 256, 0, stream>>>(W1, W1);    // range ifft
  transpose2048<<<tg, tb, 0, stream>>>(W1, W0);
  fft2048_rows<1, true, true><<<N2048, 256, 0, stream>>>(W0, W0);    // azimuth ifft
  transpose2048<<<tg, tb, 0, stream>>>(W0, W1);                      // W1 = image

  emit_kernel<<<PW_BLOCKS, 256, 0, stream>>>(W1, out_image);
}
